// MultiHeadedAttention_54735063220771
// MI455X (gfx1250) — matrix operand order repaired and checked by reading, not compiled, not run on hardware
//
#include <hip/hip_runtime.h>
#include <hip/hip_bf16.h>
#include <math.h>
#include <stdint.h>

#define B_DIM  8
#define S_DIM  1024
#define E_DIM  1024
#define H_DIM  16
#define DH_DIM 64

typedef __bf16 bf16;
typedef __attribute__((ext_vector_type(16))) __bf16 v16bf;
typedef __attribute__((ext_vector_type(8)))  __bf16 v8bf;
typedef __attribute__((ext_vector_type(4)))  __bf16 v4bf;
typedef __attribute__((ext_vector_type(8)))  float  v8f;

// ---------------------------------------------------------------------------
// WMMA helpers (CDNA5 wave32, 16x16x32 bf16 -> f32)
// ---------------------------------------------------------------------------
static __device__ __forceinline__ v8f wmma_bf16(v16bf a, v16bf b, v8f c) {
  // 8 args: (neg_a, A, neg_b, B, c_mod, C, reuse_a, reuse_b)
  return __builtin_amdgcn_wmma_f32_16x16x32_bf16(false, a, false, b, (short)0, c,
                                                 false, false);
}

// A operand, 16x32 bf16, src row-major with leading dim `ld` (elements).
// Lane l: row m = l&15; per-lane k = {16*half .. +15}
static __device__ __forceinline__ v16bf load_a_frag(const bf16* src, int ld, int lane) {
  const int m = lane & 15, half = lane >> 4;
  const bf16* p = src + (size_t)m * ld + half * 16;
  v8bf lo = *(const v8bf*)(p);
  v8bf hi = *(const v8bf*)(p + 8);
  return __builtin_shufflevector(lo, hi, 0, 1, 2, 3, 4, 5, 6, 7,
                                 8, 9, 10, 11, 12, 13, 14, 15);
}

// B operand, 32x16 bf16, column n contiguous along k: src points at (n-row, k0).
// Lane l: n = l&15, k = i + 16*half  -> single 32B load.
static __device__ __forceinline__ v16bf load_b_frag(const bf16* src, int ld, int lane) {
  const int n = lane & 15, half = lane >> 4;
  return *(const v16bf*)(src + (size_t)n * ld + half * 16);
}

// ---------------------------------------------------------------------------
// Kernel 1: f32 -> bf16 conversion (vectorized x4, grid-stride)
// ---------------------------------------------------------------------------
__global__ void cvt_f32_bf16_v4(const float4* __restrict__ src,
                                v4bf* __restrict__ dst, int n4) {
  int i = blockIdx.x * blockDim.x + threadIdx.x;
  const int stride = gridDim.x * blockDim.x;
  for (; i < n4; i += stride) {
    float4 f = src[i];
    v4bf o;
    o[0] = (bf16)f.x; o[1] = (bf16)f.y; o[2] = (bf16)f.z; o[3] = (bf16)f.w;
    dst[i] = o;
  }
}

// ---------------------------------------------------------------------------
// C-tile store helper: 16x16 f32 C tile -> bf16 (optionally transposed layout)
// ---------------------------------------------------------------------------
static __device__ __forceinline__ void store_tile_bf16(bf16* out, int bh, int mt,
                                                       int nt, v8f acc, float scale,
                                                       int transpose, int lane) {
  const int n = lane & 15, half = lane >> 4;
  if (!transpose) {
    bf16* o = out + ((size_t)bh * S_DIM + mt) * DH_DIM + nt + n;
#pragma unroll
    for (int r = 0; r < 8; ++r)
      o[(size_t)(r + 8 * half) * DH_DIM] = (bf16)(acc[r] * scale);
  } else {
    v8bf pv;
#pragma unroll
    for (int r = 0; r < 8; ++r) pv[r] = (bf16)(acc[r] * scale);
    *(v8bf*)(out + ((size_t)bh * DH_DIM + nt + n) * S_DIM + mt + 8 * half) = pv;
  }
}

// ---------------------------------------------------------------------------
// Kernel 2: per-head projection  out[b,h,s,d] = scale * sum_e X[b,s,e] W[h,d,e]
// block = 128 threads (4 waves); each wave owns a 32x32 macro-tile (4 WMMAs /
// 4 fragment loads -> 2x operand reuse).  grid = (S/64, B, H).
// transpose_out=1 stores Vt[b,h,d,s].
// ---------------------------------------------------------------------------
__global__ void qkv_proj(const bf16* __restrict__ X, const bf16* __restrict__ W,
                         bf16* __restrict__ out, float scale, int transpose_out) {
  const int lane = threadIdx.x & 31;
  const int wave = threadIdx.x >> 5;
  const int b = blockIdx.y, h = blockIdx.z;
  const int m0 = blockIdx.x * 64 + (wave & 1) * 32;  // 2 m-subtiles: m0, m0+16
  const int n0 = (wave >> 1) * 32;                   // 2 n-subtiles: n0, n0+16

  const bf16* A0 = X + ((size_t)b * S_DIM + m0) * E_DIM;
  const bf16* A1 = A0 + (size_t)16 * E_DIM;
  const bf16* B0 = W + ((size_t)h * DH_DIM + n0) * E_DIM;
  const bf16* B1 = B0 + (size_t)16 * E_DIM;

  v8f c00 = {}, c01 = {}, c10 = {}, c11 = {};
#pragma unroll 2
  for (int k0 = 0; k0 < E_DIM; k0 += 32) {
    v16bf a0 = load_a_frag(A0 + k0, E_DIM, lane);
    v16bf a1 = load_a_frag(A1 + k0, E_DIM, lane);
    v16bf b0 = load_b_frag(B0 + k0, E_DIM, lane);
    v16bf b1 = load_b_frag(B1 + k0, E_DIM, lane);
    c00 = wmma_bf16(a0, b0, c00);
    c01 = wmma_bf16(a0, b1, c01);
    c10 = wmma_bf16(a1, b0, c10);
    c11 = wmma_bf16(a1, b1, c11);
  }

  const int bh = b * H_DIM + h;
  store_tile_bf16(out, bh, m0,      n0,      c00, scale, transpose_out, lane);
  store_tile_bf16(out, bh, m0,      n0 + 16, c01, scale, transpose_out, lane);
  store_tile_bf16(out, bh, m0 + 16, n0,      c10, scale, transpose_out, lane);
  store_tile_bf16(out, bh, m0 + 16, n0 + 16, c11, scale, transpose_out, lane);
}

// ---------------------------------------------------------------------------
// Kernel 3: flash attention.  One wave owns a 16-query tile.
// Scores computed transposed (S^T = K * Q^T) so softmax reduces in-lane + 1 shfl.
// K tiles are double-buffered in LDS via global_load_async_to_lds_b128 (ASYNCcnt),
// issued one iteration ahead to overlap HBM/L2 latency with WMMA + softmax.
// ---------------------------------------------------------------------------
#define WPB 4
__global__ void flash_attn(const bf16* __restrict__ Q, const bf16* __restrict__ K,
                           const bf16* __restrict__ Vt, bf16* __restrict__ ctx) {
  __shared__ __align__(64) bf16 Plds[WPB][16][32];             // P re-layout tile
  __shared__ __align__(64) bf16 Klds[WPB][2][32][DH_DIM];      // K double buffer
  const int lane = threadIdx.x & 31;
  const int wave = threadIdx.x >> 5;
  const int bh = blockIdx.y;
  const int b = bh / H_DIM, h = bh % H_DIM;
  const int q0 = (blockIdx.x * WPB + wave) * 16;
  const int half = lane >> 4;
  const int qn = lane & 15;

  const bf16* Qp = Q + ((size_t)bh * S_DIM + q0) * DH_DIM;
  const bf16* Kp = K + (size_t)bh * S_DIM * DH_DIM;
  const bf16* Vp = Vt + (size_t)bh * DH_DIM * S_DIM;

  // Async-stage one 32x64 K tile (4 KB) into LDS buffer `buf`:
  // 8 x (32 lanes x 16B) async b128 transfers per wave.
  auto stage_k = [&](int buf, int kt) {
    const uint64_t g0 = (uint64_t)(uintptr_t)(Kp + (size_t)kt * DH_DIM);
    const uint32_t l0 = (uint32_t)(uintptr_t)&Klds[wave][buf][0][0];
#pragma unroll
    for (int i = 0; i < 8; ++i) {
      const uint32_t byte_off = (uint32_t)(i * 32 + lane) * 16u;
      const uint64_t ga = g0 + byte_off;
      const uint32_t la = l0 + byte_off;
      asm volatile("global_load_async_to_lds_b128 %0, %1, off"
                   :: "v"(la), "v"(ga) : "memory");
    }
  };

  // Q^T as B operand (n = query, k = d), K-dim 64 -> two resident fragments
  v16bf qb0 = load_b_frag(Qp + 0, DH_DIM, lane);
  v16bf qb1 = load_b_frag(Qp + 32, DH_DIM, lane);

  v8f acc0 = {}, acc1 = {}, acc2 = {}, acc3 = {};  // ctx [16q x 64d]
  float m_run = -INFINITY, l_run = 0.f;

  stage_k(0, 0);
  int buf = 0;
  for (int kt = 0; kt < S_DIM; kt += 32) {
    // prefetch next tile, then wait for the current one (<=8 leaves the new
    // batch of 8 in flight; the previous 8 — ours — are complete)
    if (kt + 32 < S_DIM) {
      stage_k(buf ^ 1, kt + 32);
      asm volatile("s_wait_asynccnt 8" ::: "memory");
    } else {
      asm volatile("s_wait_asynccnt 0" ::: "memory");
    }

    // --- scores^T: 32 keys x 16 queries = two 16x16 C tiles, K-dim = 64 ---
    const bf16* K0 = &Klds[wave][buf][0][0];
    const bf16* K1 = K0 + 16 * DH_DIM;
    v8f s0 = {}, s1 = {};
    s0 = wmma_bf16(load_a_frag(K0 + 0, DH_DIM, lane), qb0, s0);
    s0 = wmma_bf16(load_a_frag(K0 + 32, DH_DIM, lane), qb1, s0);
    s1 = wmma_bf16(load_a_frag(K1 + 0, DH_DIM, lane), qb0, s1);
    s1 = wmma_bf16(load_a_frag(K1 + 32, DH_DIM, lane), qb1, s1);

    // --- online softmax (per query column) ---
    float mx = s0[0];
#pragma unroll
    for (int r = 1; r < 8; ++r) mx = fmaxf(mx, s0[r]);
#pragma unroll
    for (int r = 0; r < 8; ++r) mx = fmaxf(mx, s1[r]);
    mx = fmaxf(mx, __shfl_xor(mx, 16, 32));  // combine the two key-halves
    const float m_new = fmaxf(m_run, mx);
    const float alpha = __expf(m_run - m_new);
    float lsum = 0.f;
    v8bf p0, p1;
#pragma unroll
    for (int r = 0; r < 8; ++r) {
      float e0 = __expf(s0[r] - m_new);
      float e1 = __expf(s1[r] - m_new);
      lsum += e0 + e1;
      p0[r] = (bf16)e0;
      p1[r] = (bf16)e1;
    }
    lsum += __shfl_xor(lsum, 16, 32);
    l_run = l_run * alpha + lsum;
    m_run = m_new;

    // --- re-layout P via LDS: Plds[q][key_local], key_local = sub*16+8*half+r ---
    *(v8bf*)&Plds[wave][qn][8 * half] = p0;
    *(v8bf*)&Plds[wave][qn][16 + 8 * half] = p1;
    asm volatile("s_wait_dscnt 0" ::: "memory");
    v16bf ap = load_a_frag(&Plds[wave][0][0], 32, lane);  // A: m=query, k=key

    // --- rescale ctx accumulators (alpha is per query row m = r+8*half) ---
#pragma unroll
    for (int r = 0; r < 8; ++r) {
      const float ar = __shfl(alpha, r + 8 * half, 32);
      acc0[r] *= ar; acc1[r] *= ar; acc2[r] *= ar; acc3[r] *= ar;
    }

    // --- ctx += P x V  (V^T stored [d][s]: contiguous along key) ---
    acc0 = wmma_bf16(ap, load_b_frag(Vp + (size_t)0 * 16 * S_DIM + kt, S_DIM, lane), acc0);
    acc1 = wmma_bf16(ap, load_b_frag(Vp + (size_t)1 * 16 * S_DIM + kt, S_DIM, lane), acc1);
    acc2 = wmma_bf16(ap, load_b_frag(Vp + (size_t)2 * 16 * S_DIM + kt, S_DIM, lane), acc2);
    acc3 = wmma_bf16(ap, load_b_frag(Vp + (size_t)3 * 16 * S_DIM + kt, S_DIM, lane), acc3);

    buf ^= 1;
  }

  // --- finalize: divide by l, store concat-head ctx [B,S,E] as bf16 ---
  bf16* o = ctx + ((size_t)b * S_DIM + q0) * E_DIM + h * DH_DIM;
#pragma unroll
  for (int r = 0; r < 8; ++r) {
    const float inv = 1.0f / __shfl(l_run, r + 8 * half, 32);
    bf16* orow = o + (size_t)(r + 8 * half) * E_DIM;
    orow[0 * 16 + qn] = (bf16)(acc0[r] * inv);
    orow[1 * 16 + qn] = (bf16)(acc1[r] * inv);
    orow[2 * 16 + qn] = (bf16)(acc2[r] * inv);
    orow[3 * 16 + qn] = (bf16)(acc3[r] * inv);
  }
}

// ---------------------------------------------------------------------------
// Kernel 4: out = ctx @ Wo^T + bo   (M=B*S, N=E, K=E), f32 output.
// 32x32 macro-tile per wave, block covers 64x64.
// ---------------------------------------------------------------------------
__global__ void out_proj(const bf16* __restrict__ A, const bf16* __restrict__ Wob,
                         const float* __restrict__ bo, float* __restrict__ out) {
  const int lane = threadIdx.x & 31;
  const int wave = threadIdx.x >> 5;
  const int m0 = blockIdx.y * 64 + (wave & 1) * 32;
  const int n0 = blockIdx.x * 64 + (wave >> 1) * 32;

  const bf16* A0 = A + (size_t)m0 * E_DIM;
  const bf16* A1 = A0 + (size_t)16 * E_DIM;
  const bf16* B0 = Wob + (size_t)n0 * E_DIM;  // Wo row n contiguous along k=e'
  const bf16* B1 = B0 + (size_t)16 * E_DIM;

  v8f c00 = {}, c01 = {}, c10 = {}, c11 = {};
#pragma unroll 2
  for (int k0 = 0; k0 < E_DIM; k0 += 32) {
    v16bf a0 = load_a_frag(A0 + k0, E_DIM, lane);
    v16bf a1 = load_a_frag(A1 + k0, E_DIM, lane);
    v16bf b0 = load_b_frag(B0 + k0, E_DIM, lane);
    v16bf b1 = load_b_frag(B1 + k0, E_DIM, lane);
    c00 = wmma_bf16(a0, b0, c00);
    c01 = wmma_bf16(a0, b1, c01);
    c10 = wmma_bf16(a1, b0, c10);
    c11 = wmma_bf16(a1, b1, c11);
  }

  const int n = lane & 15, half = lane >> 4;
  const float bias0 = bo[n0 + n];
  const float bias1 = bo[n0 + 16 + n];
#pragma unroll
  for (int r = 0; r < 8; ++r) {
    const size_t row0 = (size_t)(m0 + r + 8 * half) * E_DIM;
    const size_t row1 = (size_t)(m0 + 16 + r + 8 * half) * E_DIM;
    out[row0 + n0 + n]      = c00[r] + bias0;
    out[row0 + n0 + 16 + n] = c01[r] + bias1;
    out[row1 + n0 + n]      = c10[r] + bias0;
    out[row1 + n0 + 16 + n] = c11[r] + bias1;
  }
}

// ---------------------------------------------------------------------------
extern "C" void kernel_launch(void* const* d_in, const int* in_sizes, int n_in,
                              void* d_out, int out_size, void* d_ws, size_t ws_size,
                              hipStream_t stream) {
  (void)in_sizes; (void)n_in; (void)out_size; (void)ws_size;
  const float* Xk = (const float*)d_in[0];
  const float* Xv = (const float*)d_in[1];
  const float* Xq = (const float*)d_in[2];
  const float* Wk = (const float*)d_in[3];
  const float* Wv = (const float*)d_in[4];
  const float* Wq = (const float*)d_in[5];
  const float* Wo = (const float*)d_in[6];
  const float* bo = (const float*)d_in[7];
  float* out = (float*)d_out;

  const size_t XN = (size_t)B_DIM * S_DIM * E_DIM;           // 8 Mi elems
  const size_t WN = (size_t)H_DIM * DH_DIM * E_DIM;          // 1 Mi elems
  const size_t ON = (size_t)E_DIM * E_DIM;                   // 1 Mi elems
  const size_t QN = (size_t)B_DIM * H_DIM * S_DIM * DH_DIM;  // 8 Mi elems

  char* ws = (char*)d_ws;
  bf16* Xkb = (bf16*)ws;  ws += XN * sizeof(bf16);
  bf16* Xvb = (bf16*)ws;  ws += XN * sizeof(bf16);
  bf16* Xqb = (bf16*)ws;  ws += XN * sizeof(bf16);
  bf16* Wkb = (bf16*)ws;  ws += WN * sizeof(bf16);
  bf16* Wvb = (bf16*)ws;  ws += WN * sizeof(bf16);
  bf16* Wqb = (bf16*)ws;  ws += WN * sizeof(bf16);
  bf16* Wob = (bf16*)ws;  ws += ON * sizeof(bf16);
  bf16* Qb  = (bf16*)ws;  ws += QN * sizeof(bf16);  // [B,H,S,DH]
  bf16* Kb  = (bf16*)ws;  ws += QN * sizeof(bf16);  // [B,H,S,DH]
  bf16* Vtb = (bf16*)ws;  ws += QN * sizeof(bf16);  // [B,H,DH,S]
  bf16* Ctx = (bf16*)ws;  ws += XN * sizeof(bf16);  // [B,S,E]

  // 1) convert to bf16
  dim3 cb(256), cg(2048);
  cvt_f32_bf16_v4<<<cg, cb, 0, stream>>>((const float4*)Xk, (v4bf*)Xkb, (int)(XN / 4));
  cvt_f32_bf16_v4<<<cg, cb, 0, stream>>>((const float4*)Xv, (v4bf*)Xvb, (int)(XN / 4));
  cvt_f32_bf16_v4<<<cg, cb, 0, stream>>>((const float4*)Xq, (v4bf*)Xqb, (int)(XN / 4));
  cvt_f32_bf16_v4<<<128, cb, 0, stream>>>((const float4*)Wk, (v4bf*)Wkb, (int)(WN / 4));
  cvt_f32_bf16_v4<<<128, cb, 0, stream>>>((const float4*)Wv, (v4bf*)Wvb, (int)(WN / 4));
  cvt_f32_bf16_v4<<<128, cb, 0, stream>>>((const float4*)Wq, (v4bf*)Wqb, (int)(WN / 4));
  cvt_f32_bf16_v4<<<128, cb, 0, stream>>>((const float4*)Wo, (v4bf*)Wob, (int)(ON / 4));

  // 2) Q/K/V projections (softmax scale folded into Q; V stored transposed)
  dim3 pb(128), pg(S_DIM / 64, B_DIM, H_DIM);
  qkv_proj<<<pg, pb, 0, stream>>>(Xqb, Wqb, Qb, 0.125f, 0);
  qkv_proj<<<pg, pb, 0, stream>>>(Xkb, Wkb, Kb, 1.0f, 0);
  qkv_proj<<<pg, pb, 0, stream>>>(Xvb, Wvb, Vtb, 1.0f, 1);

  // 3) flash attention -> Ctx (concat heads, bf16)
  dim3 ab(32 * WPB), ag(S_DIM / (16 * WPB), B_DIM * H_DIM);
  flash_attn<<<ag, ab, 0, stream>>>(Qb, Kb, Vtb, Ctx);

  // 4) output projection + bias -> f32
  dim3 ob(128), og(E_DIM / 64, (B_DIM * S_DIM) / 64);
  out_proj<<<og, ob, 0, stream>>>(Ctx, Wob, bo, out);
}
